// Model_80994493268408
// MI455X (gfx1250) — matrix operand order repaired and checked by reading, not compiled, not run on hardware
//
#include <hip/hip_runtime.h>
#include <hip/hip_fp16.h>

typedef __attribute__((ext_vector_type(16))) _Float16 v16h;
typedef __attribute__((ext_vector_type(8)))  float    v8f;

constexpr int B  = 128;
constexpr int T  = 512;
constexpr int D0 = 128;
constexpr int H  = 256;
constexpr int G  = 1024;   // 4*H
constexpr float EPSC = 1e-7f;
constexpr float TEMPc = 0.1f;
constexpr float WRc = 1e-6f, BRc = 1e-6f, DRc = 1e-5f;

// ---------------------------------------------------------------- scalars
__global__ void k_prep(const float* pl1, const float* prl1, const float* pl2,
                       const float* prl2, const float* pl3, const float* prl3,
                       const float* pld, float* pv) {
  int i = threadIdx.x;
  const float* ps[7] = {pl1, prl1, pl2, prl2, pl3, prl3, pld};
  if (i < 7) {
    float x = ps[i][0];
    pv[i] = 1.f / (1.f + expf(-x));
  }
}

// ------------------------------------------------- concrete dropout masks
// stores mask(u,p)/(1-p)  (scaled retain mask)
__global__ void k_masks(const float* nx1, const float* nh1, const float* nx2,
                        const float* nh2, const float* nx3, const float* nh3,
                        const float* nd, const float* pv,
                        float* mx1, float* mh1, float* mx2, float* mh2,
                        float* mx3, float* mh3, float* md) {
  int i = blockIdx.x * blockDim.x + threadIdx.x;
  const int n1 = B * D0;   // 16384
  const int n2 = B * H;    // 32768
  const float* src; float* dst; float p; int off;
  if (i < n1) { src = nx1; dst = mx1; p = pv[0]; off = i; }
  else {
    i -= n1;
    int seg = i / n2, r = i % n2;
    if (seg >= 6) return;
    const float* S[6] = {nh1, nx2, nh2, nx3, nh3, nd};
    float* Dt[6]      = {mh1, mx2, mh2, mx3, mh3, md};
    src = S[seg]; dst = Dt[seg]; p = pv[seg + 1]; off = r;
  }
  float u  = src[off];
  float dp = logf(p + EPSC) - logf(1.f - p + EPSC)
           + logf(u + EPSC) - logf(1.f - u + EPSC);
  float m  = 1.f - 1.f / (1.f + expf(-dp / TEMPc));
  dst[off] = m / (1.f - p);
}

// ---------------------------------------------------------- x -> fp16
__global__ void k_cvtx(const float* __restrict__ x, _Float16* __restrict__ dst, int n) {
  int i = blockIdx.x * 256 + threadIdx.x;
  if (i < n) dst[i] = (_Float16)x[i];
}

// ----------------------- fuse [Wih|Whh] into WMMA B-fragment-ready blob
// blob index = (((nt*KT + kt)*32 + L)*16 + e)
// value = W[n][k],  n = nt*16 + (L&15),  k = kt*32 + (L>=16 ? 16 : 0) + e
__global__ void k_convw(const float* __restrict__ Wih, const float* __restrict__ Whh,
                        int Din, int KT, _Float16* __restrict__ dst) {
  int idx = blockIdx.x * 256 + threadIdx.x;
  int total = 64 * KT * 512;
  if (idx >= total) return;
  int e  = idx & 15;
  int L  = (idx >> 4) & 31;
  int kt = (idx >> 9) % KT;
  int nt = idx / (512 * KT);
  int n = nt * 16 + (L & 15);
  int k = kt * 32 + ((L >= 16) ? 16 : 0) + e;
  float v = (k < Din) ? Wih[n * Din + k] : Whh[n * H + (k - Din)];
  dst[idx] = (_Float16)v;
}

// ---------------------------------------------------------- bias fusion
__global__ void k_bias(const float* a1, const float* b1, const float* a2,
                       const float* b2, const float* a3, const float* b3,
                       float* d1, float* d2, float* d3) {
  int i = blockIdx.x * 256 + threadIdx.x;
  if (i >= G) return;
  d1[i] = a1[i] + b1[i];
  d2[i] = a2[i] + b2[i];
  d3[i] = a3[i] + b3[i];
}

// -------------------------------------- deterministic sum-of-squares x14
__global__ void k_sumsq(const float* t0, const float* t1, const float* t2,
                        const float* t3, const float* t4, const float* t5,
                        const float* t6, const float* t7, const float* t8,
                        const float* t9, const float* t10, const float* t11,
                        const float* t12, const float* t13, float* out) {
  const float* ts[14] = {t0,t1,t2,t3,t4,t5,t6,t7,t8,t9,t10,t11,t12,t13};
  const int ns[14] = {G*D0, G*H, G, G,  G*H, G*H, G, G,  G*H, G*H, G, G,  7*H, 7};
  __shared__ float red[256];
  int b = blockIdx.x;
  const float* src = ts[b];
  int n = ns[b];
  float s = 0.f;
  for (int i = threadIdx.x; i < n; i += 256) { float v = src[i]; s += v * v; }
  red[threadIdx.x] = s;
  __syncthreads();
  for (int st = 128; st > 0; st >>= 1) {
    if ((int)threadIdx.x < st) red[threadIdx.x] += red[threadIdx.x + st];
    __syncthreads();
  }
  if (threadIdx.x == 0) out[b] = red[0];
}

// ---------------------------------------------------------------- LSTM
// One workgroup = 16 batch rows, full time loop, no cross-WG sync.
// Gates GEMM: A [16 x K] (xm_t | hm) in LDS (pre-swizzled A-frag layout),
// B = weight blob in fragment layout, 8 waves x 8 n-tiles x (K/32) k-tiles
// of v_wmma_f32_16x16x32_f16.
__global__ __launch_bounds__(256)
void k_lstm(const _Float16* __restrict__ xseq, int Din, int K, int KT,
            const _Float16* __restrict__ wblob, const float* __restrict__ bias,
            const float* __restrict__ mx, const float* __restrict__ mh,
            _Float16* __restrict__ outSeq, float* __restrict__ hlast) {
  __shared__ __align__(32) _Float16 Asw[16 * 512];   // 16 KB, swizzled A frags
  __shared__ float hbuf[16 * H];                     // 16 KB
  extern __shared__ float gates[];                   // 16*G f32 = 64 KB dynamic

  const int tid  = threadIdx.x;
  const int lane = tid & 31;
  const int w    = tid >> 5;
  const int b0   = blockIdx.x * 16;

  for (int i = tid; i < 16 * H; i += 256) hbuf[i] = 0.f;
  float creg[16];
#pragma unroll
  for (int u = 0; u < 16; ++u) creg[u] = 0.f;
  __syncthreads();

  const v8f vzero = {0.f, 0.f, 0.f, 0.f, 0.f, 0.f, 0.f, 0.f};
  const int AK = 16 * K;
  const _Float16* wb = wblob + (size_t)(w * 8) * KT * 512;  // per-wave N-slice

  for (int t = 0; t < T; ++t) {
    // (a) build swizzled A tile: [xm_t | hm] -> fp16 fragment layout
    for (int idx = tid; idx < AK; idx += 256) {
      int m = idx / K, k = idx - m * K;
      float v;
      if (k < Din) {
        v = (float)xseq[((size_t)(b0 + m) * T + t) * Din + k] * mx[(b0 + m) * Din + k];
      } else {
        int hh = k - Din;
        v = hbuf[m * H + hh] * mh[(b0 + m) * H + hh];
      }
      int kt = k >> 5, r = k & 31;
      int hi = (r >> 4) & 1, rr = r & 15;
      int lsel, e;
      if (hi == 0) { lsel = m;      e = rr; }
      else         { lsel = m + 16; e = rr; }
      Asw[((kt * 32 + lsel) << 4) + e] = (_Float16)v;
    }
    __syncthreads();

    // (b) WMMA gate GEMM: this wave covers gates [w*128, w*128+128)
    v8f acc[8];
#pragma unroll
    for (int nt = 0; nt < 8; ++nt) acc[nt] = vzero;
    for (int kt = 0; kt < KT; ++kt) {
      v16h af = *(const v16h*)(&Asw[(kt * 32 + lane) << 4]);
#pragma unroll
      for (int nt = 0; nt < 8; ++nt) {
        v16h bf = *(const v16h*)(wb + (((size_t)nt * KT + kt) * 32 + lane) * 16);
        acc[nt] = __builtin_amdgcn_wmma_f32_16x16x32_f16(
            false, af, false, bf, (short)0, acc[nt], false, false);
      }
    }

    // (c) bias + scatter gates to LDS (C/D layout: M = r + 8*(lane>=16), N = lane&15)
    {
      int mbase = (lane >= 16) ? 8 : 0;
      int nloc  = lane & 15;
#pragma unroll
      for (int nt = 0; nt < 8; ++nt) {
        int g = w * 128 + nt * 16 + nloc;
        float bv = bias[g];
#pragma unroll
        for (int r = 0; r < 8; ++r)
          gates[(r + mbase) * G + g] = acc[nt][r] + bv;
      }
    }
    __syncthreads();

    // (d) LSTM cell, c in registers, h to LDS (+ sequence output)
#pragma unroll
    for (int u = 0; u < 16; ++u) {
      int j = u * 256 + tid;
      int m = j >> 8, hh = j & 255;
      float gi = gates[m * G + hh];
      float gf = gates[m * G + hh + 256];
      float gg = gates[m * G + hh + 512];
      float go = gates[m * G + hh + 768];
      float si = 1.f / (1.f + expf(-gi));
      float sf = 1.f / (1.f + expf(-gf));
      float so = 1.f / (1.f + expf(-go));
      float cn = sf * creg[u] + si * tanhf(gg);
      float hn = so * tanhf(cn);
      creg[u] = cn;
      hbuf[m * H + hh] = hn;
      if (outSeq) outSeq[((size_t)(b0 + m) * T + t) * H + hh] = (_Float16)hn;
    }
    __syncthreads();
  }

  if (hlast) {
    for (int i = tid; i < 16 * H; i += 256) {
      int m = i / H, hh = i - m * H;
      hlast[(b0 + m) * H + hh] = hbuf[i];
    }
  }
}

// ------------------------------------------------------------------ head
__global__ void k_head(const float* __restrict__ h3, const float* __restrict__ md,
                       const float* __restrict__ Wmu, const float* __restrict__ bmu,
                       const float* __restrict__ pv, const float* __restrict__ ssq,
                       float* __restrict__ out) {
  __shared__ float alpha_s[B * 7];
  int tid = threadIdx.x;
  for (int task = tid; task < B * 7; task += 256) {
    int b = task / 7, j = task - b * 7;
    float s = 0.f;
    for (int hh = 0; hh < H; ++hh)
      s += h3[b * H + hh] * md[b * H + hh] * Wmu[j * H + hh];
    s += bmu[j];
    float sp = (s > 20.f) ? s : log1pf(expf(s));
    alpha_s[task] = sp + 1.f;
  }
  __syncthreads();
  for (int b = tid; b < B; b += 256) {
    float S = 0.f;
    for (int j = 0; j < 7; ++j) S += alpha_s[b * 7 + j];
    for (int j = 0; j < 7; ++j) out[b * 7 + j] = alpha_s[b * 7 + j] / S;
    out[B * 7 + b] = S;
  }
  if (tid == 0) {
    float p1 = pv[0], pr1 = pv[1], p2 = pv[2], pr2 = pv[3];
    float p3 = pv[4], pr3 = pv[5], pd = pv[6];
    auto ent = [](float p) { return p * logf(p) + (1.f - p) * logf(1.f - p); };
    float r1 = WRc * (ssq[0] / (1.f - p1) + ssq[1] / (1.f - pr1))
             + BRc * (ssq[2] + ssq[3])
             + DRc * (128.f * ent(p1) + 256.f * ent(pr1));
    float r2 = WRc * (ssq[4] / (1.f - p2) + ssq[5] / (1.f - pr2))
             + BRc * (ssq[6] + ssq[7])
             + DRc * (256.f * ent(p2) + 256.f * ent(pr2));
    float r3 = WRc * (ssq[8] / (1.f - p3) + ssq[9] / (1.f - pr3))
             + BRc * (ssq[10] + ssq[11])
             + DRc * (256.f * ent(p3) + 256.f * ent(pr3));
    float r4 = WRc * ssq[12] / (1.f - pd) + BRc * ssq[13]
             + DRc * 256.f * ent(pd);
    out[B * 7 + B] = r1 + r2 + r3 + r4;
  }
}

// ------------------------------------------------------------------ host
extern "C" void kernel_launch(void* const* d_in, const int* in_sizes, int n_in,
                              void* d_out, int out_size, void* d_ws, size_t ws_size,
                              hipStream_t stream) {
  const float* x    = (const float*)d_in[0];
  const float* Wih1 = (const float*)d_in[1];
  const float* Whh1 = (const float*)d_in[2];
  const float* bih1 = (const float*)d_in[3];
  const float* bhh1 = (const float*)d_in[4];
  const float* Wih2 = (const float*)d_in[5];
  const float* Whh2 = (const float*)d_in[6];
  const float* bih2 = (const float*)d_in[7];
  const float* bhh2 = (const float*)d_in[8];
  const float* Wih3 = (const float*)d_in[9];
  const float* Whh3 = (const float*)d_in[10];
  const float* bih3 = (const float*)d_in[11];
  const float* bhh3 = (const float*)d_in[12];
  const float* Wmu  = (const float*)d_in[13];
  const float* bmu  = (const float*)d_in[14];
  const float* pl1  = (const float*)d_in[15];
  const float* prl1 = (const float*)d_in[16];
  const float* pl2  = (const float*)d_in[17];
  const float* prl2 = (const float*)d_in[18];
  const float* pl3  = (const float*)d_in[19];
  const float* prl3 = (const float*)d_in[20];
  const float* pld  = (const float*)d_in[21];
  const float* nx1  = (const float*)d_in[22];
  const float* nh1  = (const float*)d_in[23];
  const float* nx2  = (const float*)d_in[24];
  const float* nh2  = (const float*)d_in[25];
  const float* nx3  = (const float*)d_in[26];
  const float* nh3  = (const float*)d_in[27];
  const float* nd   = (const float*)d_in[28];
  float* out = (float*)d_out;

  char* ws = (char*)d_ws;
  size_t off = 0;
  auto take = [&](size_t bytes) {
    size_t r = off;
    off += (bytes + 255) & ~(size_t)255;
    return r;
  };
  _Float16* xf16 = (_Float16*)(ws + take((size_t)B * T * D0 * 2));
  _Float16* act1 = (_Float16*)(ws + take((size_t)B * T * H * 2));
  _Float16* act2 = (_Float16*)(ws + take((size_t)B * T * H * 2));
  _Float16* wb1  = (_Float16*)(ws + take((size_t)G * 384 * 2));
  _Float16* wb2  = (_Float16*)(ws + take((size_t)G * 512 * 2));
  _Float16* wb3  = (_Float16*)(ws + take((size_t)G * 512 * 2));
  float* bias1 = (float*)(ws + take(G * 4));
  float* bias2 = (float*)(ws + take(G * 4));
  float* bias3 = (float*)(ws + take(G * 4));
  float* mx1 = (float*)(ws + take((size_t)B * D0 * 4));
  float* mh1 = (float*)(ws + take((size_t)B * H * 4));
  float* mx2 = (float*)(ws + take((size_t)B * H * 4));
  float* mh2 = (float*)(ws + take((size_t)B * H * 4));
  float* mx3 = (float*)(ws + take((size_t)B * H * 4));
  float* mh3 = (float*)(ws + take((size_t)B * H * 4));
  float* md  = (float*)(ws + take((size_t)B * H * 4));
  float* pv  = (float*)(ws + take(64));
  float* ssq = (float*)(ws + take(64));
  float* hlast = (float*)(ws + take((size_t)B * H * 4));
  (void)ws_size; (void)n_in; (void)in_sizes; (void)out_size;

  // scalar p's
  k_prep<<<1, 32, 0, stream>>>(pl1, prl1, pl2, prl2, pl3, prl3, pld, pv);

  // masks
  {
    int total = B * D0 + 6 * B * H;
    k_masks<<<(total + 255) / 256, 256, 0, stream>>>(
        nx1, nh1, nx2, nh2, nx3, nh3, nd, pv,
        mx1, mh1, mx2, mh2, mx3, mh3, md);
  }

  // x -> fp16
  {
    int n = B * T * D0;
    k_cvtx<<<(n + 255) / 256, 256, 0, stream>>>(x, xf16, n);
  }

  // weight blobs (fragment-ready fp16)
  k_convw<<<(64 * 12 * 512 + 255) / 256, 256, 0, stream>>>(Wih1, Whh1, D0, 12, wb1);
  k_convw<<<(64 * 16 * 512 + 255) / 256, 256, 0, stream>>>(Wih2, Whh2, H, 16, wb2);
  k_convw<<<(64 * 16 * 512 + 255) / 256, 256, 0, stream>>>(Wih3, Whh3, H, 16, wb3);

  k_bias<<<4, 256, 0, stream>>>(bih1, bhh1, bih2, bhh2, bih3, bhh3, bias1, bias2, bias3);

  k_sumsq<<<14, 256, 0, stream>>>(Wih1, Whh1, bih1, bhh1, Wih2, Whh2, bih2, bhh2,
                                  Wih3, Whh3, bih3, bhh3, Wmu, bmu, ssq);

  // recurrent layers (8 WGs = 8 independent batch tiles of 16)
  const int dynLds = 16 * G * 4;  // 64 KB gate buffer
  (void)hipFuncSetAttribute((const void*)k_lstm,
                            hipFuncAttributeMaxDynamicSharedMemorySize, dynLds);
  k_lstm<<<8, 256, dynLds, stream>>>(xf16, D0, 384, 12, wb1, bias1, mx1, mh1,
                                     act1, (float*)nullptr);
  k_lstm<<<8, 256, dynLds, stream>>>(act1, H, 512, 16, wb2, bias2, mx2, mh2,
                                     act2, (float*)nullptr);
  k_lstm<<<8, 256, dynLds, stream>>>(act2, H, 512, 16, wb3, bias3, mx3, mh3,
                                     (_Float16*)nullptr, hlast);

  // head: prob [128,7], S [128], reg [1]
  k_head<<<1, 256, 0, stream>>>(hlast, md, Wmu, bmu, pv, ssq, out);
}
